// ToothFormer_8091718386280
// MI455X (gfx1250) — matrix operand order repaired and checked by reading, not compiled, not run on hardware
//
#include <hip/hip_runtime.h>
#include <cmath>

#define B_   4
#define N_   16384
#define M_   256
#define E_   256
#define D_   6
#define FF_  512
#define C_   10
#define H_   8
#define HD_  32
#define K_   128
#define LN_EPS 1e-5f

typedef __attribute__((ext_vector_type(16))) __bf16 v16bf;
typedef __attribute__((ext_vector_type(8)))  __bf16 v8bf;
typedef __attribute__((ext_vector_type(8)))  float  v8f;

// pack two f32 -> bf16x2 in one dword
__device__ inline unsigned int pack_bf16x2(float a, float b) {
  union { __bf16 h[2]; unsigned int u; } x;
  x.h[0] = (__bf16)a; x.h[1] = (__bf16)b;
  return x.u;
}

// ---------------- WMMA fragment helpers (ISA 7.12.2 layouts, wave32) ----------------
// All loads are 16-byte contiguous chunks -> ds_load_b128.

// A fragment: 16x32 bf16, source row-major, leading dim `ld` (multiple of 8).
// lane l: row r = l&15; elements 0..7 = K 8*hi..8*hi+7 ; elements 8..15 = K 16+8*hi..+7
__device__ inline v16bf frag_a_bf16(const __bf16* base, int ld, int lane) {
  int r = lane & 15, hi = lane >> 4;
  const __bf16* row = base + (size_t)r * ld + 8 * hi;
  v8bf a0 = *(const v8bf*)(row);
  v8bf a1 = *(const v8bf*)(row + 16);
  return __builtin_shufflevector(a0, a1, 0, 1, 2, 3, 4, 5, 6, 7,
                                         8, 9, 10, 11, 12, 13, 14, 15);
}

// B fragment with logical B[k][n] = src[n][k]; src row-major [N][K], ld = K stride
// lane l: col n = l&15; elements 0..7 = K 8*hi..8*hi+7 ; elements 8..15 = K 16+8*hi..+7
__device__ inline v16bf frag_bt_bf16(const __bf16* base, int ld, int lane) {
  int n = lane & 15, hi = lane >> 4;
  const __bf16* row = base + (size_t)n * ld + 8 * hi;
  v8bf b0 = *(const v8bf*)(row);
  v8bf b1 = *(const v8bf*)(row + 16);
  return __builtin_shufflevector(b0, b1, 0, 1, 2, 3, 4, 5, 6, 7,
                                         8, 9, 10, 11, 12, 13, 14, 15);
}

__device__ inline v8f wmma_bf16(v16bf a, v16bf b, v8f c) {
  return __builtin_amdgcn_wmma_f32_16x16x32_bf16(false, a, false, b, (short)0, c, false, false);
}

// ---------------- centers ----------------
__global__ void k_centers(const float* __restrict__ xyz, float* __restrict__ centers,
                          float* __restrict__ csq) {
  int i = blockIdx.x * blockDim.x + threadIdx.x;
  if (i >= B_ * M_) return;
  int b = i / M_, m = i % M_;
  float t = (float)m * ((float)(N_ - 1) / (float)(M_ - 1));
  int ci = (int)t;
  const float* p = xyz + ((size_t)b * N_ + ci) * 3;
  float x = p[0], y = p[1], z = p[2];
  centers[i * 3] = x; centers[i * 3 + 1] = y; centers[i * 3 + 2] = z;
  csq[i] = x * x + y * y + z * z;
}

// ---------------- kNN: per-center LDS radix select of 128 smallest distances ----------------
__global__ void k_knn(const float* __restrict__ xyz, const float* __restrict__ centers,
                      int* __restrict__ idx_knn) {
  extern __shared__ char dynsm[];
  unsigned int* keys = (unsigned int*)dynsm;        // N_ keys (64 KB)
  unsigned int* hist = keys + N_;                   // 256 bins
  unsigned int* ctrl = hist + 256;                  // [0]=selbin [1]=cnt< [2]=cnt== [3]=kth
  int bm = blockIdx.x;
  int b  = bm / M_;
  int t  = threadIdx.x;
  const float* c = centers + (size_t)bm * 3;
  float cx = c[0], cy = c[1], cz = c[2];
  const float* P = xyz + (size_t)b * N_ * 3;
  for (int p = t; p < N_; p += 256) {
    float px = P[p * 3], py = P[p * 3 + 1], pz = P[p * 3 + 2];
    float d = px * px + py * py + pz * pz - 2.f * (cx * px + cy * py + cz * pz);
    unsigned int u = __float_as_uint(d);
    u = (u & 0x80000000u) ? ~u : (u | 0x80000000u);   // monotone map
    keys[p] = u;
  }
  if (t == 0) ctrl[3] = K_;
  __syncthreads();

  unsigned int prefix = 0, pmask = 0;
  for (int shift = 24; shift >= 0; shift -= 8) {
    hist[t] = 0;
    __syncthreads();
    for (int p = t; p < N_; p += 256) {
      unsigned int u = keys[p];
      if ((u & pmask) == prefix) atomicAdd(&hist[(u >> shift) & 255u], 1u);
    }
    __syncthreads();
    if (t == 0) {
      unsigned int kth = ctrl[3], cum = 0, bsel = 255;
      for (int bq = 0; bq < 256; ++bq) {
        unsigned int cq = hist[bq];
        if (cum + cq >= kth) { bsel = (unsigned int)bq; break; }
        cum += cq;
      }
      ctrl[0] = bsel;
      ctrl[3] = kth - cum;
    }
    __syncthreads();
    prefix |= ctrl[0] << shift;
    pmask  |= 255u << shift;
  }
  unsigned int T = prefix;
  if (t == 0) { ctrl[1] = 0; ctrl[2] = 0; }
  __syncthreads();
  int* out = idx_knn + (size_t)bm * K_;
  for (int p = t; p < N_; p += 256)
    if (keys[p] < T) { unsigned int s = atomicAdd(&ctrl[1], 1u); out[s] = p; }
  __syncthreads();
  unsigned int base = ctrl[1];
  for (int p = t; p < N_; p += 256)
    if (keys[p] == T) {
      unsigned int s = base + atomicAdd(&ctrl[2], 1u);
      if (s < K_) out[s] = p;
    }
}

// ---------------- neighbor conv (3->64->128->256) + maxpool + pe ----------------
__global__ void k_conv(const float* __restrict__ xyz, const float* __restrict__ centers,
                       const int* __restrict__ idx_knn,
                       const float* __restrict__ w1, const float* __restrict__ b1,
                       const float* __restrict__ w2, const float* __restrict__ b2,
                       const float* __restrict__ w3, const float* __restrict__ b3,
                       const float* __restrict__ pe, float* __restrict__ tokens) {
  extern __shared__ char dynsm[];
  float*  local = (float*)dynsm;                                   // 128*3 f32 (1536 B)
  __bf16* f1    = (__bf16*)(dynsm + 128 * 3 * 4);                  // [128][64]
  __bf16* f2    = (__bf16*)((char*)f1 + 128 * 64 * 2);             // [128][128]
  __bf16* W2n   = (__bf16*)((char*)f2 + 128 * 128 * 2);            // [128 o][64 k]
  __bf16* W3n   = (__bf16*)((char*)W2n + 128 * 64 * 2);            // [256 o][128 k]
  int bm = blockIdx.x;
  int b = bm / M_, m = bm % M_;
  int t = threadIdx.x, wv = t >> 5, lane = t & 31;

  // stage weights in natural [o][k] layout (bf16, packed-pair stores)
  for (int i = t; i < 128 * 32; i += 256) {
    int o = i >> 5, k2 = (i & 31) * 2;
    const float* wp = w2 + (size_t)o * 64 + k2;
    *(unsigned int*)(W2n + (size_t)o * 64 + k2) = pack_bf16x2(wp[0], wp[1]);
  }
  for (int i = t; i < 256 * 64; i += 256) {
    int o = i >> 6, k2 = (i & 63) * 2;
    const float* wp = w3 + (size_t)o * 128 + k2;
    *(unsigned int*)(W3n + (size_t)o * 128 + k2) = pack_bf16x2(wp[0], wp[1]);
  }
  const float* c = centers + (size_t)bm * 3;
  float cx = c[0], cy = c[1], cz = c[2];
  if (t < K_) {
    int pi = idx_knn[(size_t)bm * K_ + t];
    const float* p = xyz + ((size_t)b * N_ + pi) * 3;
    local[t * 3]     = p[0] - cx;
    local[t * 3 + 1] = p[1] - cy;
    local[t * 3 + 2] = p[2] - cz;
  }
  __syncthreads();

  // stage 1 (K=3, VALU): f1 = relu(local @ w1^T + b1), pair-packed stores
  for (int i = t; i < K_ * 32; i += 256) {
    int k = i >> 5, o = (i & 31) * 2;
    float l0 = local[k * 3], l1 = local[k * 3 + 1], l2 = local[k * 3 + 2];
    float a0 = b1[o]     + w1[o * 3] * l0     + w1[o * 3 + 1] * l1     + w1[o * 3 + 2] * l2;
    float a1 = b1[o + 1] + w1[(o+1) * 3] * l0 + w1[(o+1) * 3 + 1] * l1 + w1[(o+1) * 3 + 2] * l2;
    *(unsigned int*)(f1 + (size_t)k * 64 + o) = pack_bf16x2(fmaxf(a0, 0.f), fmaxf(a1, 0.f));
  }
  __syncthreads();

  // stage 2 (WMMA): f2 = relu(f1[128x64] @ w2^T + b2); wave w -> M-tile w
  {
    v8f acc[8] = {};
    for (int ks = 0; ks < 2; ++ks) {
      v16bf a = frag_a_bf16(f1 + (size_t)(wv * 16) * 64 + ks * 32, 64, lane);
#pragma unroll
      for (int nt = 0; nt < 8; ++nt) {
        v16bf bb = frag_bt_bf16(W2n + (size_t)(nt * 16) * 64 + ks * 32, 64, lane);
        acc[nt] = wmma_bf16(a, bb, acc[nt]);
      }
    }
    int n = lane & 15, hi = lane >> 4;
    for (int nt = 0; nt < 8; ++nt) {
      int col = nt * 16 + n;
      float bias = b2[col];
#pragma unroll
      for (int v = 0; v < 8; ++v) {
        int row = wv * 16 + v + 8 * hi;
        f2[(size_t)row * 128 + col] = (__bf16)fmaxf(acc[nt][v] + bias, 0.f);
      }
    }
  }
  __syncthreads();

  // stage 3 (WMMA) + maxpool over K: wave w handles N-tiles {w, w+8}
  {
    v8f mx[2];
#pragma unroll
    for (int j = 0; j < 2; ++j)
#pragma unroll
      for (int v = 0; v < 8; ++v) mx[j][v] = -3.0e38f;
    for (int mt = 0; mt < 8; ++mt) {
      v8f acc[2] = {};
      for (int ks = 0; ks < 4; ++ks) {
        v16bf a = frag_a_bf16(f2 + (size_t)(mt * 16) * 128 + ks * 32, 128, lane);
#pragma unroll
        for (int j = 0; j < 2; ++j) {
          int nt = wv + 8 * j;
          v16bf bb = frag_bt_bf16(W3n + (size_t)(nt * 16) * 128 + ks * 32, 128, lane);
          acc[j] = wmma_bf16(a, bb, acc[j]);
        }
      }
#pragma unroll
      for (int j = 0; j < 2; ++j)
#pragma unroll
        for (int v = 0; v < 8; ++v) mx[j][v] = fmaxf(mx[j][v], acc[j][v]);
    }
    int n = lane & 15, hi = lane >> 4;
#pragma unroll
    for (int j = 0; j < 2; ++j) {
      float v0 = mx[j][0];
#pragma unroll
      for (int v = 1; v < 8; ++v) v0 = fmaxf(v0, mx[j][v]);
      v0 = fmaxf(v0, __shfl_xor(v0, 16, 32));
      if (hi == 0) {
        int col = (wv + 8 * j) * 16 + n;
        tokens[(size_t)bm * E_ + col] = v0 + b3[col] + pe[(size_t)m * E_ + col];
      }
    }
  }
}

// ---------------- LayerNorm (one row per block, blockDim = E_) ----------------
__global__ void k_ln(const float* __restrict__ x, const float* __restrict__ g,
                     const float* __restrict__ bta, float* __restrict__ y) {
  __shared__ float red[E_];
  int row = blockIdx.x, t = threadIdx.x;
  float v = x[(size_t)row * E_ + t];
  red[t] = v; __syncthreads();
  for (int s = E_ / 2; s > 0; s >>= 1) { if (t < s) red[t] += red[t + s]; __syncthreads(); }
  float mu = red[0] * (1.f / E_);
  __syncthreads();
  float d = v - mu;
  red[t] = d * d; __syncthreads();
  for (int s = E_ / 2; s > 0; s >>= 1) { if (t < s) red[t] += red[t + s]; __syncthreads(); }
  float var = red[0] * (1.f / E_);
  y[(size_t)row * E_ + t] = d * rsqrtf(var + LN_EPS) * g[t] + bta[t];
}

// ---------------- generic WMMA GEMM: C = [resid +] [relu](A[rows,K] @ W^T + bias) ----------------
// W row-major [cols][K] (kept in that layout in LDS -> contiguous B-fragment loads).
__global__ void k_gemm(const float* __restrict__ A, const float* __restrict__ W,
                       const float* __restrict__ bias, const float* __restrict__ resid,
                       float* __restrict__ Cout, int rows, int K, int cols, int relu) {
  __shared__ __align__(16) __bf16 As[128 * 64];   // [128 rows][64 k]
  __shared__ __align__(16) __bf16 Bs[128 * 64];   // [128 cols][64 k]
  int t = threadIdx.x, wv = t >> 5, lane = t & 31;
  int m0 = blockIdx.y * 128, n0 = blockIdx.x * 128;
  v8f acc[8] = {};
  for (int k0 = 0; k0 < K; k0 += 64) {
    for (int i = t; i < 128 * 32; i += 256) {
      int r = i >> 5, cc = (i & 31) * 2;
      int gm = m0 + r;
      float2 v2;
      if (gm < rows) v2 = *(const float2*)(A + (size_t)gm * K + k0 + cc);
      else { v2.x = 0.f; v2.y = 0.f; }
      *(unsigned int*)(As + (size_t)r * 64 + cc) = pack_bf16x2(v2.x, v2.y);
    }
    for (int i = t; i < 128 * 32; i += 256) {
      int n = i >> 5, kk = (i & 31) * 2;
      int gn = n0 + n;
      float2 v2;
      if (gn < cols) v2 = *(const float2*)(W + (size_t)gn * K + k0 + kk);
      else { v2.x = 0.f; v2.y = 0.f; }
      *(unsigned int*)(Bs + (size_t)n * 64 + kk) = pack_bf16x2(v2.x, v2.y);
    }
    __syncthreads();
    for (int ks = 0; ks < 2; ++ks) {
      v16bf a = frag_a_bf16(As + (size_t)(wv * 16) * 64 + ks * 32, 64, lane);
#pragma unroll
      for (int nt = 0; nt < 8; ++nt) {
        v16bf bb = frag_bt_bf16(Bs + (size_t)(nt * 16) * 64 + ks * 32, 64, lane);
        acc[nt] = wmma_bf16(a, bb, acc[nt]);
      }
    }
    __syncthreads();
  }
  int n = lane & 15, hi = lane >> 4;
  for (int nt = 0; nt < 8; ++nt) {
    int col = n0 + nt * 16 + n;
    if (col >= cols) continue;
    float bv = bias ? bias[col] : 0.f;
#pragma unroll
    for (int v = 0; v < 8; ++v) {
      int row = m0 + wv * 16 + v + 8 * hi;
      if (row >= rows) continue;
      float val = acc[nt][v] + bv;
      if (relu) val = fmaxf(val, 0.f);
      if (resid) val += resid[(size_t)row * cols + col];
      Cout[(size_t)row * cols + col] = val;
    }
  }
}

// ---------------- attention: one block per (b, h); full 256x256 scores via WMMA ----------------
__global__ void k_attn(const float* __restrict__ qkv, float* __restrict__ obuf) {
  extern __shared__ char dynsm[];
  __bf16* Qs = (__bf16*)dynsm;          // [256 m][32 d], pre-scaled by 1/sqrt(hd)
  __bf16* Ks = Qs + 256 * 32;           // [256 m][32 d]
  __bf16* Vt = Ks + 256 * 32;           // [32 d][256 m]  (transposed for B-frags)
  __bf16* Al = Vt + 32 * 256;           // per-wave prob tiles: 8 * [16][256]
  int bh = blockIdx.x;
  int b = bh / H_, h = bh % H_;
  int t = threadIdx.x, wv = t >> 5, lane = t & 31;
  const float scale = 0.17677669529663687f;   // 1/sqrt(32)
  const float* base = qkv + (size_t)b * M_ * (3 * E_);
  for (int i = t; i < M_ * HD_ / 2; i += 256) {
    int m = i >> 4, d = (i & 15) * 2;
    const float* rowp = base + (size_t)m * (3 * E_);
    float2 q2 = *(const float2*)(rowp + h * HD_ + d);
    *(unsigned int*)(Qs + (size_t)m * 32 + d) = pack_bf16x2(q2.x * scale, q2.y * scale);
    float2 k2 = *(const float2*)(rowp + E_ + h * HD_ + d);
    *(unsigned int*)(Ks + (size_t)m * 32 + d) = pack_bf16x2(k2.x, k2.y);
    float2 v2 = *(const float2*)(rowp + 2 * E_ + h * HD_ + d);
    Vt[(size_t)d * 256 + m]       = (__bf16)v2.x;
    Vt[(size_t)(d + 1) * 256 + m] = (__bf16)v2.y;
  }
  __syncthreads();
  __bf16* aw = Al + (size_t)wv * (16 * 256);
  int nL = lane & 15, hi = lane >> 4;
  for (int half = 0; half < 2; ++half) {
    int mt = wv * 2 + half;   // 16 M-tiles over 8 waves
    v16bf qa = frag_a_bf16(Qs + (size_t)(mt * 16) * HD_, HD_, lane);
    v8f s[16];
#pragma unroll
    for (int nt = 0; nt < 16; ++nt) {
      v16bf kb = frag_bt_bf16(Ks + (size_t)(nt * 16) * HD_, HD_, lane);   // K^T, hd==32
      v8f z = {};
      s[nt] = wmma_bf16(qa, kb, z);
    }
    // row-wise softmax: row m = v + 8*hi; columns spread over 16 tiles x 16 lanes
#pragma unroll
    for (int v = 0; v < 8; ++v) {
      float mx = s[0][v];
#pragma unroll
      for (int nt = 1; nt < 16; ++nt) mx = fmaxf(mx, s[nt][v]);
      for (int off = 8; off >= 1; off >>= 1) mx = fmaxf(mx, __shfl_xor(mx, off, 32));
      float e[16], sum = 0.f;
#pragma unroll
      for (int nt = 0; nt < 16; ++nt) { e[nt] = expf(s[nt][v] - mx); sum += e[nt]; }
      for (int off = 8; off >= 1; off >>= 1) sum += __shfl_xor(sum, off, 32);
      float inv = 1.f / sum;
      int row = v + 8 * hi;
#pragma unroll
      for (int nt = 0; nt < 16; ++nt)
        aw[(size_t)row * 256 + nt * 16 + nL] = (__bf16)(e[nt] * inv);
    }
    __syncthreads();
    v8f o0 = {}, o1 = {};
#pragma unroll
    for (int ks = 0; ks < 8; ++ks) {
      v16bf aa = frag_a_bf16(aw + ks * 32, 256, lane);
      v16bf bv0 = frag_bt_bf16(Vt + 0 * 256 + ks * 32, 256, lane);     // d-cols 0..15
      o0 = wmma_bf16(aa, bv0, o0);
      v16bf bv1 = frag_bt_bf16(Vt + (size_t)16 * 256 + ks * 32, 256, lane); // d-cols 16..31
      o1 = wmma_bf16(aa, bv1, o1);
    }
    float* out = obuf + ((size_t)b * M_) * E_ + h * HD_;
#pragma unroll
    for (int v = 0; v < 8; ++v) {
      int row = mt * 16 + v + 8 * hi;
      out[(size_t)row * E_ + nL]      = o0[v];
      out[(size_t)row * E_ + 16 + nL] = o1[v];
    }
    __syncthreads();
  }
}

// ---------------- argmin over centers + gather of token-level outputs ----------------
__global__ void k_argmin_gather(const float* __restrict__ xyz, const float* __restrict__ centers,
                                const float* __restrict__ csq, const float* __restrict__ out_tok,
                                float* __restrict__ out) {
  __shared__ float cs[M_ * 3];
  __shared__ float cq[M_];
  int t = threadIdx.x;
  int blocksPerB = N_ / 256;
  int b  = blockIdx.x / blocksPerB;
  int n0 = (blockIdx.x % blocksPerB) * 256;
  for (int i = t; i < M_ * 3; i += 256) cs[i] = centers[(size_t)b * M_ * 3 + i];
  cq[t] = csq[b * M_ + t];
  __syncthreads();
  int n = n0 + t;
  const float* p = xyz + ((size_t)b * N_ + n) * 3;
  float px = p[0], py = p[1], pz = p[2];
  float best = 3.4e38f; int bi = 0;
  for (int m = 0; m < M_; ++m) {
    float d = cq[m] - 2.f * (px * cs[m * 3] + py * cs[m * 3 + 1] + pz * cs[m * 3 + 2]);
    if (d < best) { best = d; bi = m; }
  }
  const float* src = out_tok + ((size_t)b * M_ + bi) * C_;
  float* dst = out + ((size_t)b * N_ + n) * C_;
#pragma unroll
  for (int c = 0; c < C_; ++c) dst[c] = src[c];
}

// ---------------- host orchestration ----------------
extern "C" void kernel_launch(void* const* d_in, const int* in_sizes, int n_in,
                              void* d_out, int out_size, void* d_ws, size_t ws_size,
                              hipStream_t stream) {
  (void)in_sizes; (void)n_in; (void)out_size; (void)ws_size;
  const float* xyz     = (const float*)d_in[0];
  const float* conv_w1 = (const float*)d_in[1];
  const float* conv_b1 = (const float*)d_in[2];
  const float* conv_w2 = (const float*)d_in[3];
  const float* conv_b2 = (const float*)d_in[4];
  const float* conv_w3 = (const float*)d_in[5];
  const float* conv_b3 = (const float*)d_in[6];
  const float* pe      = (const float*)d_in[7];
  const float* enc_wqkv  = (const float*)d_in[8];
  const float* enc_bqkv  = (const float*)d_in[9];
  const float* enc_wo    = (const float*)d_in[10];
  const float* enc_bo    = (const float*)d_in[11];
  const float* enc_ln1_g = (const float*)d_in[12];
  const float* enc_ln1_b = (const float*)d_in[13];
  const float* enc_ln2_g = (const float*)d_in[14];
  const float* enc_ln2_b = (const float*)d_in[15];
  const float* enc_w1    = (const float*)d_in[16];
  const float* enc_b1    = (const float*)d_in[17];
  const float* enc_w2    = (const float*)d_in[18];
  const float* enc_b2    = (const float*)d_in[19];
  const float* proj_w  = (const float*)d_in[20];
  const float* proj_b  = (const float*)d_in[21];
  const float* head_w1 = (const float*)d_in[22];
  const float* head_b1 = (const float*)d_in[23];
  const float* head_w2 = (const float*)d_in[24];
  const float* head_b2 = (const float*)d_in[25];

  char* ws = (char*)d_ws;
  size_t off = 0;
  auto alloc = [&](size_t bytes) -> char* {
    char* p = ws + off;
    off += (bytes + 255) & ~(size_t)255;
    return p;
  };
  float* centers = (float*)alloc((size_t)B_ * M_ * 3 * 4);
  float* csq     = (float*)alloc((size_t)B_ * M_ * 4);
  int*   idx_knn = (int*)  alloc((size_t)B_ * M_ * K_ * 4);
  float* xbuf    = (float*)alloc((size_t)B_ * M_ * E_ * 4);       // token stream
  float* lnb     = (float*)alloc((size_t)B_ * M_ * E_ * 4);
  float* qkv     = (float*)alloc((size_t)B_ * M_ * 3 * E_ * 4);
  float* ob      = (float*)alloc((size_t)B_ * M_ * E_ * 4);
  float* ff      = (float*)alloc((size_t)B_ * M_ * FF_ * 4);
  float* tok_out = (float*)alloc((size_t)B_ * M_ * C_ * 4);

  k_centers<<<(B_ * M_ + 255) / 256, 256, 0, stream>>>(xyz, centers, csq);

  size_t knn_lds = (size_t)N_ * 4 + 256 * 4 + 16;
  k_knn<<<B_ * M_, 256, knn_lds, stream>>>(xyz, centers, idx_knn);

  size_t conv_lds = 128 * 3 * 4 + 128 * 64 * 2 + 128 * 128 * 2 + 128 * 64 * 2 + 256 * 128 * 2;
  k_conv<<<B_ * M_, 256, conv_lds, stream>>>(xyz, centers, idx_knn,
      conv_w1, conv_b1, conv_w2, conv_b2, conv_w3, conv_b3, pe, xbuf);

  size_t attn_lds = (size_t)3 * 256 * 32 * 2 + (size_t)8 * 16 * 256 * 2;
  for (int l = 0; l < D_; ++l) {
    const float* wqkv = enc_wqkv + (size_t)l * 3 * E_ * E_;
    const float* bqkv = enc_bqkv + (size_t)l * 3 * E_;
    const float* wo   = enc_wo   + (size_t)l * E_ * E_;
    const float* bo   = enc_bo   + (size_t)l * E_;
    const float* g1   = enc_ln1_g + (size_t)l * E_;
    const float* b1l  = enc_ln1_b + (size_t)l * E_;
    const float* g2   = enc_ln2_g + (size_t)l * E_;
    const float* b2l  = enc_ln2_b + (size_t)l * E_;
    const float* w1   = enc_w1 + (size_t)l * FF_ * E_;
    const float* bb1  = enc_b1 + (size_t)l * FF_;
    const float* w2   = enc_w2 + (size_t)l * E_ * FF_;
    const float* bb2  = enc_b2 + (size_t)l * E_;

    k_ln<<<B_ * M_, E_, 0, stream>>>(xbuf, g1, b1l, lnb);
    k_gemm<<<dim3(6, 8), 256, 0, stream>>>(lnb, wqkv, bqkv, nullptr, qkv,
                                           B_ * M_, E_, 3 * E_, 0);
    k_attn<<<B_ * H_, 256, attn_lds, stream>>>(qkv, ob);
    k_gemm<<<dim3(2, 8), 256, 0, stream>>>(ob, wo, bo, xbuf, xbuf,
                                           B_ * M_, E_, E_, 0);        // x += attn_out
    k_ln<<<B_ * M_, E_, 0, stream>>>(xbuf, g2, b2l, lnb);
    k_gemm<<<dim3(4, 8), 256, 0, stream>>>(lnb, w1, bb1, nullptr, ff,
                                           B_ * M_, E_, FF_, 1);       // relu
    k_gemm<<<dim3(2, 8), 256, 0, stream>>>(ff, w2, bb2, xbuf, xbuf,
                                           B_ * M_, FF_, E_, 0);       // x += ffn
  }

  // token-level head MLP (factorized: per-point MLP depends only on gathered token)
  k_gemm<<<dim3(2, 8), 256, 0, stream>>>(xbuf, proj_w, proj_b, nullptr, lnb,
                                         B_ * M_, E_, E_, 0);
  k_gemm<<<dim3(2, 8), 256, 0, stream>>>(lnb, head_w1, head_b1, nullptr, ob,
                                         B_ * M_, E_, E_, 1);
  k_gemm<<<dim3(1, 8), 256, 0, stream>>>(ob, head_w2, head_b2, nullptr, tok_out,
                                         B_ * M_, E_, C_, 0);

  k_argmin_gather<<<B_ * (N_ / 256), 256, 0, stream>>>(xyz, centers, csq, tok_out,
                                                       (float*)d_out);
}
